// neural_network_equivariant_15625091022902
// MI455X (gfx1250) — compile-verified, matrix operand order checked by reading, not run on hardware
//
#include <hip/hip_runtime.h>
#include <hip/hip_bf16.h>
#include <math.h>

// ---------------------------------------------------------------------------
// Equivariant GNN forward, MI455X (gfx1250, wave32, WMMA).
// Radial MLP (E x 16 -> 64 -> 64 -> 96) runs on v_wmma_f32_16x16x32_f16,
// fused with edge geometry + message scatter (f32 global atomics).
// LDS weight/activation buffers are stored in WMMA-fragment-native layout so
//每 operand load is contiguous (ds_load_b128) instead of 16x ds_load_u16.
// ---------------------------------------------------------------------------

typedef __attribute__((ext_vector_type(16))) _Float16 v16h;
typedef __attribute__((ext_vector_type(8)))  float    v8f;

#define PI_F 3.14159265358979f
#define INV_SQRT_NN 0.17677669529663687f   // 1/sqrt(32)
#define SQRT3_F 1.7320508075688772f
#define BESSEL_C 1.7888543819998317f       // sqrt(2/5)*sqrt(8)

__device__ __forceinline__ float fast_rcp(float x) {
  return __builtin_amdgcn_rcpf(x);
}
__device__ __forceinline__ float silu_f(float x) {
  return x * fast_rcp(1.0f + __expf(-x));
}
__device__ __forceinline__ float sigmoid_f(float x) {
  return fast_rcp(1.0f + __expf(-x));
}

__device__ __forceinline__ v8f wmma_f16(v16h a, v16h b, v8f c) {
  return __builtin_amdgcn_wmma_f32_16x16x32_f16(false, a, false, b, (short)0, c, false, false);
}

// WMMA 16-bit fragment half-index -> local K offset (within a 32-K block):
//   k = 16*(i>=8) + 8*g + (i&7)      (g = lane>>4)
__device__ __forceinline__ int frag_k(int i, int g) {
  return ((i >= 8) ? 16 : 0) + 8 * g + (i & 7);
}

// Activation staging in A-fragment physical order.
// Element (m, col):  lane = m + 16*col[3];  within-lane = 16*col[5] + 8*col[4] + col[2:0]
// Lane region is 32 halves: [frag kbase=0 | frag kbase=32], each contiguous v16h.
__device__ __forceinline__ int hstage_colpart(int col) {
  return ((col >> 5) << 4) + (((col >> 4) & 1) << 3) + (col & 7);
}

// ---------------------------------------------------------------------------
__global__ void init_nodes_kernel(const float* __restrict__ x,
                                  const int* __restrict__ node_attr,
                                  const float* __restrict__ embed,
                                  const float* __restrict__ Wu,
                                  float* __restrict__ xv, float* __restrict__ y_s,
                                  float* __restrict__ y_v, float* __restrict__ yo_s,
                                  float* __restrict__ yo_v, float* __restrict__ emb_g,
                                  int N) {
  int tid = blockIdx.x * blockDim.x + threadIdx.x;
  int n = tid / 48, c = tid % 48;
  if (n >= N) return;
  if (c < 6) xv[n * 6 + c] = x[n * 6 + c];
  if (c < 16) {
    int a = node_attr[n];
    a = a < 0 ? 0 : (a > 19 ? 19 : a);
    emb_g[n * 16 + c] = embed[a * 16 + c];
#pragma unroll
    for (int d = 0; d < 3; ++d) {
      float v = x[n * 6 + d] * Wu[c] + x[n * 6 + 3 + d] * Wu[16 + c];
      y_v[n * 48 + c * 3 + d] = v;
      yo_v[n * 48 + c * 3 + d] = v;
    }
  } else {
    int u = c - 16;
    y_s[n * 32 + u] = 0.0f;
    yo_s[n * 32 + u] = 0.0f;
  }
}

// ---------------------------------------------------------------------------
__global__ void node_pre_kernel(const float* __restrict__ y_s,
                                const float* __restrict__ y_v,
                                const float* __restrict__ W1s,
                                const float* __restrict__ W1v,
                                float* __restrict__ s1, float* __restrict__ v1,
                                float* __restrict__ agg_s, float* __restrict__ agg_v,
                                int N) {
  int tid = blockIdx.x * blockDim.x + threadIdx.x;
  int n = tid / 192, c = tid % 192;
  if (n >= N) return;
  agg_v[n * 192 + c] = 0.0f;
  if (c < 32) {
    agg_s[n * 32 + c] = 0.0f;
    float acc = 0.0f;
#pragma unroll
    for (int k = 0; k < 32; ++k) acc += y_s[n * 32 + k] * W1s[k * 32 + c];
    s1[n * 32 + c] = acc;
  }
  if (c < 16) {
#pragma unroll
    for (int d = 0; d < 3; ++d) {
      float acc = 0.0f;
#pragma unroll
      for (int k = 0; k < 16; ++k) acc += y_v[n * 48 + k * 3 + d] * W1v[k * 16 + c];
      v1[n * 48 + c * 3 + d] = acc;
    }
  }
}

// ---------------------------------------------------------------------------
// Fused edge kernel. 128 threads = 4 waves, 16 edges per wave per chunk.
// ---------------------------------------------------------------------------
__global__ __launch_bounds__(128) void edge_kernel(
    const float* __restrict__ xv, const int* __restrict__ esrcG,
    const int* __restrict__ edstG, const float* __restrict__ s1g,
    const float* __restrict__ v1g, const float* __restrict__ Wr0,
    const float* __restrict__ br0, const float* __restrict__ Wr1,
    const float* __restrict__ br1, const float* __restrict__ Wr2,
    const float* __restrict__ br2, float* __restrict__ agg_s,
    float* __restrict__ agg_v, int E) {
  // B-operand weights stored directly as WMMA fragments: [tile][lane] -> v16h
  __shared__ v16h bfragWr0[4][32];          // 4 KB   (K padded 16->32)
  __shared__ v16h bfragWr1[2][4][32];       // 8 KB   (2 K-blocks x 4 N-tiles)
  __shared__ v16h bfragWr2[2][6][32];       // 12 KB  (2 K-blocks x 6 N-tiles)
  __shared__ float sbias[64 + 64 + 96];
  // activation staging in A-fragment physical order (per wave)
  __shared__ v16h hstageA[4][64];           // 8 KB  (lane*2 + kblock)
  __shared__ v16h hstageB[4][64];           // 8 KB
  __shared__ float wbuf[4][16 * 96];        // 24 KB
  __shared__ float aab[4][16][6];
  __shared__ int   esrc[4][16];
  __shared__ int   edst[4][16];

  const int lane = threadIdx.x & 31;
  const int wv   = threadIdx.x >> 5;
  const int g    = lane >> 4;
  const int m    = lane & 15;

  // ---- cooperative weight load, swizzled to fragment layout (once/block)
  {
    _Float16* f0 = (_Float16*)bfragWr0;
    for (int idx = threadIdx.x; idx < 4 * 32 * 16; idx += 128) {
      int t = idx >> 9, l = (idx >> 4) & 31, i = idx & 15;
      int n = (l & 15) + 16 * t;
      int k = frag_k(i, l >> 4);
      f0[idx] = (_Float16)(k < 16 ? Wr0[k * 64 + n] : 0.0f);
    }
    _Float16* f1 = (_Float16*)bfragWr1;
    for (int idx = threadIdx.x; idx < 2 * 4 * 32 * 16; idx += 128) {
      int kk = idx >> 11, t = (idx >> 9) & 3, l = (idx >> 4) & 31, i = idx & 15;
      int n = (l & 15) + 16 * t;
      int k = kk * 32 + frag_k(i, l >> 4);
      f1[idx] = (_Float16)Wr1[k * 64 + n];
    }
    _Float16* f2 = (_Float16*)bfragWr2;
    for (int idx = threadIdx.x; idx < 2 * 6 * 32 * 16; idx += 128) {
      int kk = idx / 3072, rem = idx % 3072;
      int t = rem >> 9, l = (rem >> 4) & 31, i = idx & 15;
      int n = (l & 15) + 16 * t;
      int k = kk * 32 + frag_k(i, l >> 4);
      f2[idx] = (_Float16)Wr2[k * 96 + n];
    }
    for (int i = threadIdx.x; i < 224; i += 128)
      sbias[i] = (i < 64) ? br0[i] : (i < 128 ? br1[i - 64] : br2[i - 128]);
  }
  __syncthreads();

  _Float16* HA = (_Float16*)hstageA[wv];
  _Float16* HB = (_Float16*)hstageB[wv];

  const int nChunks = (E + 63) >> 6;
  for (int chunk = blockIdx.x; chunk < nChunks; chunk += gridDim.x) {
    const int ebase = chunk * 64 + wv * 16;
    const int eIdx = ebase + m;
    const int ce = eIdx < E ? eIdx : (E - 1);

    // ---- phase 1: per-edge geometry + bessel (lane group g handles row g)
    const int s = esrcG[ce];
    const int d_ = edstG[ce];
    if (g == 0) { esrc[wv][m] = s; edst[wv][m] = d_; }
    float e0 = xv[s * 6 + g * 3 + 0] - xv[d_ * 6 + g * 3 + 0];
    float e1 = xv[s * 6 + g * 3 + 1] - xv[d_ * 6 + g * 3 + 1];
    float e2 = xv[s * 6 + g * 3 + 2] - xv[d_ * 6 + g * 3 + 2];
    float r = sqrtf(e0 * e0 + e1 * e1 + e2 * e2);
    float rinv = fast_rcp(r + 1e-9f);
    float u = 2.0f * (r * 0.2f - 1.0f);
    float cut = (u > 0.0f) ? 0.0f
                           : ((u < -1.0f) ? 1.0f : (1.0f - __cosf(PI_F * u)) * 0.5f);
    float cs = cut * SQRT3_F * rinv;
    aab[wv][m][g * 3 + 0] = cs * e0;
    aab[wv][m][g * 3 + 1] = cs * e1;
    aab[wv][m][g * 3 + 2] = cs * e2;

    // bessel features straight into the A-fragment layout:
    // half i (<8) is k = 8*g + i (n = i+1); upper K half zero (padding).
    v16h afeat;
#pragma unroll
    for (int i = 0; i < 8; ++i)
      afeat[i] = (_Float16)(BESSEL_C * __sinf((float)(i + 1) * PI_F * r * 0.2f) * rinv);
#pragma unroll
    for (int i = 8; i < 16; ++i) afeat[i] = (_Float16)0.0f;

    // ---- GEMM1: feat(16x32 padded) @ Wr0 -> silu -> hstageA
#pragma unroll
    for (int t = 0; t < 4; ++t) {
      v8f c = {};
      c = wmma_f16(afeat, bfragWr0[t][lane], c);
      const int col = m + 16 * t;
      const int lcol = ((col >> 3) & 1) << 4;
      const int cpart = hstage_colpart(col);
      const float bias = sbias[col];
#pragma unroll
      for (int j = 0; j < 8; ++j)
        HA[(j + 8 * g + lcol) * 32 + cpart] = (_Float16)silu_f(c[j] + bias);
    }

    // ---- GEMM2: h0 @ Wr1 -> silu -> hstageB  (A frags are contiguous v16h)
    {
      v16h a0 = hstageA[wv][lane * 2 + 0];
      v16h a1 = hstageA[wv][lane * 2 + 1];
#pragma unroll
      for (int t = 0; t < 4; ++t) {
        v8f c = {};
        c = wmma_f16(a0, bfragWr1[0][t][lane], c);
        c = wmma_f16(a1, bfragWr1[1][t][lane], c);
        const int col = m + 16 * t;
        const int lcol = ((col >> 3) & 1) << 4;
        const int cpart = hstage_colpart(col);
        const float bias = sbias[64 + col];
#pragma unroll
        for (int j = 0; j < 8; ++j)
          HB[(j + 8 * g + lcol) * 32 + cpart] = (_Float16)silu_f(c[j] + bias);
      }
    }

    // ---- GEMM3: h1 @ Wr2 -> wbuf (f32, +bias)
    {
      v16h a0 = hstageB[wv][lane * 2 + 0];
      v16h a1 = hstageB[wv][lane * 2 + 1];
      float* Wb = wbuf[wv];
#pragma unroll
      for (int t = 0; t < 6; ++t) {
        v8f c = {};
        c = wmma_f16(a0, bfragWr2[0][t][lane], c);
        c = wmma_f16(a1, bfragWr2[1][t][lane], c);
        const int col = m + 16 * t;
        const float bias = sbias[128 + col];
#pragma unroll
        for (int j = 0; j < 8; ++j)
          Wb[(j + 8 * g) * 96 + col] = c[j] + bias;
      }
    }

    // ---- message phase: lane = channel u (0..31), loop over the 16 edges
    const int uch = lane;
    const int up = uch & 15;
    for (int e = 0; e < 16; ++e) {
      const int eI = ebase + e;
      if (eI >= E) break;                 // uniform across wave
      const int ss = esrc[wv][e];
      const int dd = edst[wv][e];
      const float a0x = aab[wv][e][0], a0y = aab[wv][e][1], a0z = aab[wv][e][2];
      const float a1x = aab[wv][e][3], a1y = aab[wv][e][4], a1z = aab[wv][e][5];
      const float vx = v1g[ss * 48 + up * 3 + 0];
      const float vy = v1g[ss * 48 + up * 3 + 1];
      const float vz = v1g[ss * 48 + up * 3 + 2];
      const float dval = (uch < 16) ? (vx * a0x + vy * a0y + vz * a0z)
                                    : (vx * a1x + vy * a1y + vz * a1z);
      const float* we = &wbuf[wv][e * 96];
      atomicAdd(&agg_s[dd * 32 + uch], we[64 + uch] * dval * INV_SQRT_NN);
      const float s1u = s1g[ss * 32 + uch];
      const float b0 = we[uch] * s1u * INV_SQRT_NN;
      const float b1 = we[32 + uch] * s1u * INV_SQRT_NN;
      float* gv = &agg_v[(size_t)dd * 192];
      atomicAdd(gv + uch * 3 + 0, b0 * a0x);
      atomicAdd(gv + uch * 3 + 1, b0 * a0y);
      atomicAdd(gv + uch * 3 + 2, b0 * a0z);
      atomicAdd(gv + 96 + uch * 3 + 0, b1 * a1x);
      atomicAdd(gv + 96 + uch * 3 + 1, b1 * a1y);
      atomicAdd(gv + 96 + uch * 3 + 2, b1 * a1z);
    }
  }
}

// ---------------------------------------------------------------------------
__global__ __launch_bounds__(256) void node_update_kernel(
    float* __restrict__ y_s, float* __restrict__ y_v, float* __restrict__ yo_s,
    float* __restrict__ yo_v, float* __restrict__ xv,
    const float* __restrict__ agg_s, const float* __restrict__ agg_v,
    const float* __restrict__ emb_g, const float* __restrict__ W2s,
    const float* __restrict__ W2v, const float* __restrict__ Wsc_s,
    const float* __restrict__ Wsc_v, const float* __restrict__ Wsi_s,
    const float* __restrict__ Wsi_v, const float* __restrict__ Wp,
    const float* __restrict__ hArr, const float* __restrict__ mixArr,
    int layer, int N) {
  __shared__ float cs[4][48];
  __shared__ float nyv[4][48];
  const int slot = threadIdx.x >> 6;
  const int t = threadIdx.x & 63;
  const int n = blockIdx.x * 4 + slot;
  const bool act = (n < N);

  const float hv = hArr[layer];
  const float dt = fminf(fmaxf(hv * hv, 1e-4f), 0.1f);
  const float mp = mixArr[layer];
  const float mix = fminf(mp * mp, 1.0f);

  if (act && t < 48) {
    float acc = 0.0f;
#pragma unroll
    for (int u = 0; u < 32; ++u) acc += agg_s[n * 32 + u] * W2s[u * 48 + t];
    for (int u = 0; u < 32; ++u) {
      const float ys = y_s[n * 32 + u];
#pragma unroll
      for (int e = 0; e < 16; ++e)
        acc += ys * emb_g[n * 16 + e] * Wsc_s[(u * 16 + e) * 48 + t];
    }
    cs[slot][t] = acc;
  }
  float cv[3] = {0.0f, 0.0f, 0.0f};
  if (act && t < 16) {
#pragma unroll
    for (int d = 0; d < 3; ++d) {
      float acc = 0.0f;
      for (int u = 0; u < 64; ++u) acc += agg_v[n * 192 + u * 3 + d] * W2v[u * 16 + t];
      for (int u = 0; u < 16; ++u) {
        const float yv = y_v[n * 48 + u * 3 + d];
#pragma unroll
        for (int e = 0; e < 16; ++e)
          acc += yv * emb_g[n * 16 + e] * Wsc_v[(u * 16 + e) * 16 + t];
      }
      cv[d] = acc;
    }
  }
  __syncthreads();

  // compute updates with OLD y values, then barrier, then write
  float ns = 0.0f, curs = 0.0f, olds = 0.0f;
  if (act && t < 32) {
    const float gs = silu_f(cs[slot][t]);
    float si = 0.0f;
#pragma unroll
    for (int k = 0; k < 32; ++k) si += y_s[n * 32 + k] * Wsi_s[k * 32 + t];
    ns = mix * gs + (1.0f - mix) * si;
    curs = y_s[n * 32 + t];
    olds = yo_s[n * 32 + t];
  }
  float nv[3] = {0, 0, 0}, curv[3] = {0, 0, 0}, oldv[3] = {0, 0, 0};
  if (act && t < 16) {
    const float gate = sigmoid_f(cs[slot][32 + t]);
#pragma unroll
    for (int d = 0; d < 3; ++d) {
      float si = 0.0f;
#pragma unroll
      for (int k = 0; k < 16; ++k) si += y_v[n * 48 + k * 3 + d] * Wsi_v[k * 16 + t];
      nv[d] = mix * gate * cv[d] + (1.0f - mix) * si;
      curv[d] = y_v[n * 48 + t * 3 + d];
      oldv[d] = yo_v[n * 48 + t * 3 + d];
    }
  }
  __syncthreads();
  if (act && t < 32) {
    y_s[n * 32 + t] = 2.0f * curs - olds + dt * ns;
    yo_s[n * 32 + t] = curs;
  }
  if (act && t < 16) {
#pragma unroll
    for (int d = 0; d < 3; ++d) {
      const float ny = 2.0f * curv[d] - oldv[d] + dt * nv[d];
      y_v[n * 48 + t * 3 + d] = ny;
      yo_v[n * 48 + t * 3 + d] = curv[d];
      nyv[slot][t * 3 + d] = ny;
    }
  }
  __syncthreads();
  if (act && t < 6) {
    const int k = t / 3, d = t % 3;
    float acc = 0.0f;
#pragma unroll
    for (int uu = 0; uu < 16; ++uu) acc += nyv[slot][uu * 3 + d] * Wp[uu * 2 + k];
    xv[n * 6 + k * 3 + d] = acc;
  }
}

__global__ void finalize_kernel(const float* __restrict__ xv, float* __restrict__ out,
                                int N) {
  int tid = blockIdx.x * blockDim.x + threadIdx.x;
  if (tid < N * 6) out[tid] = xv[tid];
  if (tid == 0) {
    out[N * 6 + 0] = -1.0f;
    out[N * 6 + 1] = -1.0f;
    out[N * 6 + 2] = 0.0f;
  }
}

// ---------------------------------------------------------------------------
extern "C" void kernel_launch(void* const* d_in, const int* in_sizes, int n_in,
                              void* d_out, int out_size, void* d_ws, size_t ws_size,
                              hipStream_t stream) {
  const float* x        = (const float*)d_in[0];
  const int*   node_attr= (const int*)d_in[2];
  const int*   edge_src = (const int*)d_in[3];
  const int*   edge_dst = (const int*)d_in[4];
  const float* embed    = (const float*)d_in[5];
  const float* Wu       = (const float*)d_in[6];
  const float* Wp       = (const float*)d_in[7];
  const float* W1s      = (const float*)d_in[8];
  const float* W1v      = (const float*)d_in[9];
  const float* Wr0      = (const float*)d_in[10];
  const float* br0      = (const float*)d_in[11];
  const float* Wr1      = (const float*)d_in[12];
  const float* br1      = (const float*)d_in[13];
  const float* Wr2      = (const float*)d_in[14];
  const float* br2      = (const float*)d_in[15];
  const float* W2s      = (const float*)d_in[16];
  const float* W2v      = (const float*)d_in[17];
  const float* Wsc_s    = (const float*)d_in[18];
  const float* Wsc_v    = (const float*)d_in[19];
  const float* Wsi_s    = (const float*)d_in[20];
  const float* Wsi_v    = (const float*)d_in[21];
  const float* hArr     = (const float*)d_in[22];
  const float* mixArr   = (const float*)d_in[23];

  const int N = in_sizes[0] / 6;
  const int E = in_sizes[3];

  float* p = (float*)d_ws;
  float* y_s   = p; p += (size_t)N * 32;
  float* yo_s  = p; p += (size_t)N * 32;
  float* y_v   = p; p += (size_t)N * 48;
  float* yo_v  = p; p += (size_t)N * 48;
  float* xv    = p; p += (size_t)N * 6;
  float* s1    = p; p += (size_t)N * 32;
  float* v1    = p; p += (size_t)N * 48;
  float* agg_s = p; p += (size_t)N * 32;
  float* agg_v = p; p += (size_t)N * 192;
  float* emb_g = p; p += (size_t)N * 16;

  init_nodes_kernel<<<(N * 48 + 255) / 256, 256, 0, stream>>>(
      x, node_attr, embed, Wu, xv, y_s, y_v, yo_s, yo_v, emb_g, N);

  const int nChunks = (E + 63) / 64;
  const int edgeGrid = nChunks < 2048 ? nChunks : 2048;

  for (int layer = 0; layer < 2; ++layer) {
    node_pre_kernel<<<(N * 192 + 255) / 256, 256, 0, stream>>>(
        y_s, y_v, W1s + layer * 32 * 32, W1v + layer * 16 * 16, s1, v1, agg_s,
        agg_v, N);

    edge_kernel<<<edgeGrid, 128, 0, stream>>>(
        xv, edge_src, edge_dst, s1, v1,
        Wr0 + layer * 16 * 64, br0 + layer * 64,
        Wr1 + layer * 64 * 64, br1 + layer * 64,
        Wr2 + layer * 64 * 96, br2 + layer * 96,
        agg_s, agg_v, E);

    node_update_kernel<<<(N + 3) / 4, 256, 0, stream>>>(
        y_s, y_v, yo_s, yo_v, xv, agg_s, agg_v, emb_g,
        W2s + layer * 32 * 48, W2v + layer * 64 * 16,
        Wsc_s + layer * 32 * 16 * 48, Wsc_v + layer * 16 * 16 * 16,
        Wsi_s + layer * 32 * 32, Wsi_v + layer * 16 * 16, Wp, hArr, mixArr,
        layer, N);
  }

  finalize_kernel<<<(N * 6 + 255) / 256, 256, 0, stream>>>(xv, (float*)d_out, N);
}
